// TrackedAttention_19662360281263
// MI455X (gfx1250) — matrix operand order repaired and checked by reading, not compiled, not run on hardware
//
#include <hip/hip_runtime.h>

// ---------------------------------------------------------------------------
// MI455X (gfx1250, wave32) fused attention:
//   K1: QKV projection GEMM (f16 WMMA, f32 accum) + bias + RoPE -> f16 ws
//   K2: flash-attention with async-LDS K/V staging (ASYNCcnt path),
//       QK^T WMMA, online softmax, P.V WMMA -> f32 out
// Shapes: B=2, S=4096, H=512, nh=8, hd=64.  ~82 GFLOP vs ~50MB traffic:
// compute bound -> matrix math on v_wmma_f32_16x16x32_f16; K/V tiles staged
// once per block via global_load_async_to_lds_b128 (4 waves share them).
// ---------------------------------------------------------------------------

typedef __attribute__((ext_vector_type(16))) _Float16 v16h;
typedef __attribute__((ext_vector_type(8)))  float    v8f;

#define NH    8
#define HD    64
#define SQ    4096
#define HDIM  512
#define NB    2

static __device__ __forceinline__ v8f wmma_f16(v16h a, v16h b, v8f c) {
  // (neg_a, A, neg_b, B, c_mod, C, reuse_a, reuse_b)
  return __builtin_amdgcn_wmma_f32_16x16x32_f16(false, a, false, b,
                                                (short)0, c, false, false);
}

// CDNA5 async LDS copy: per-lane 16B global -> LDS, tracked by ASYNCcnt.
static __device__ __forceinline__ void async_copy_b128(unsigned lds_off,
                                                       const void* gaddr) {
  asm volatile("global_load_async_to_lds_b128 %0, %1, off"
               :: "v"(lds_off), "v"(gaddr) : "memory");
}
static __device__ __forceinline__ void wait_async0() {
  asm volatile("s_wait_asynccnt 0" ::: "memory");
}

// ---------------------------------------------------------------------------
// Kernel 1: Y = X*W + b for one projection; wave computes a 16x64 tile
// (one full head worth of columns), applies RoPE in-register, stores f16
// in (b, h, s, d) layout for the attention kernel.
// grid = (M/64, 8 heads), block = 128 (4 waves)
// ---------------------------------------------------------------------------
__global__ void qkv_rope_kernel(const float* __restrict__ X,
                                const float* __restrict__ W,
                                const float* __restrict__ bias,
                                _Float16* __restrict__ Out,
                                int apply_rope) {
  const int lane  = threadIdx.x & 31;
  const int wave  = threadIdx.x >> 5;
  const int mtile = blockIdx.x * 4 + wave;   // 16-row tile, 0..511
  const int h     = blockIdx.y;              // head 0..7
  const int half  = lane >> 4;
  const int col   = lane & 15;

  v8f acc[4] = {};                            // 16x64 f32 accumulators
  const int arow = mtile * 16 + col;          // A-fragment row for this lane

  for (int kb = 0; kb < HDIM; kb += 32) {
    // A fragment: X rows, f32 -> f16 (A layout: K = 2i + 16*half)
    v16h af;
#pragma unroll
    for (int i = 0; i < 8; ++i) {
      int kk = kb + 2 * i + 16 * half;
      float2 x2 = *(const float2*)(X + (size_t)arow * HDIM + kk);
      af[2 * i]     = (_Float16)x2.x;
      af[2 * i + 1] = (_Float16)x2.y;
    }
#pragma unroll
    for (int t = 0; t < 4; ++t) {
      // B fragment: W[k][n], B layout: K = 2i + 16*half, N = col
      int n = h * HD + 16 * t + col;
      v16h bf;
#pragma unroll
      for (int i = 0; i < 8; ++i) {
        int kk = kb + 2 * i + 16 * half;
        bf[2 * i]     = (_Float16)W[(size_t)kk * HDIM + n];
        bf[2 * i + 1] = (_Float16)W[(size_t)(kk + 1) * HDIM + n];
      }
      acc[t] = wmma_f16(af, bf, acc[t]);
    }
  }

  // bias
#pragma unroll
  for (int t = 0; t < 4; ++t) {
    float bval = bias[h * HD + 16 * t + col];
#pragma unroll
    for (int r = 0; r < 8; ++r) acc[t][r] += bval;
  }

  // RoPE: pair (d, d+32) lives in (frag t, frag t+2) at the same lane/VGPR.
  if (apply_rope) {
#pragma unroll
    for (int t = 0; t < 2; ++t) {
      int dA = 16 * t + col;            // 0..31
      int dB = dA + 32;
      float invA = __powf(10000.f, -(float)(dA & ~1) / 64.f);
      float invB = __powf(10000.f, -(float)(dB & ~1) / 64.f);
#pragma unroll
      for (int r = 0; r < 8; ++r) {
        int   mg  = mtile * 16 + r + 8 * half;
        float pos = (float)(mg & (SQ - 1));
        float sA, cA, sB, cB;
        __sincosf(pos * invA, &sA, &cA);
        __sincosf(pos * invB, &sB, &cB);
        float x1 = acc[t][r], x2 = acc[t + 2][r];
        acc[t][r]     = x1 * cA - x2 * sA;   // d < 32:  x*cos - x[d+32]*sin
        acc[t + 2][r] = x2 * cB + x1 * sB;   // d >= 32: x*cos + x[d-32]*sin
      }
    }
  }

  // store f16 (b, h, s, d)
#pragma unroll
  for (int t = 0; t < 4; ++t) {
    int d = 16 * t + col;
#pragma unroll
    for (int r = 0; r < 8; ++r) {
      int mg = mtile * 16 + r + 8 * half;
      int b  = mg >> 12;                 // /4096
      int s  = mg & (SQ - 1);
      Out[(((size_t)(b * NH + h) * SQ) + s) * HD + d] = (_Float16)acc[t][r];
    }
  }
}

// ---------------------------------------------------------------------------
// Kernel 2: flash attention. One wave = one 16-query tile; 4 waves/block
// share one (b,h), so K/V 32x64 tiles are staged into LDS ONCE per block
// via async copies (ASYNCcnt), then all fragments are built from LDS.
// grid = (S/64, B*nh), block = 128 (4 waves).
// ---------------------------------------------------------------------------
__global__ void attn_kernel(const _Float16* __restrict__ Qh,
                            const _Float16* __restrict__ Kh,
                            const _Float16* __restrict__ Vh,
                            const float* __restrict__ mask,
                            float* __restrict__ Out) {
  __shared__ __align__(16) _Float16 ldsK[32 * HD];      // 4 KB
  __shared__ __align__(16) _Float16 ldsV[32 * HD];      // 4 KB
  __shared__ __align__(16) _Float16 ldsP[4][16 * 32];   // 4 KB (per-wave P)

  const int tid   = threadIdx.x;
  const int lane  = tid & 31;
  const int wave  = tid >> 5;
  const int bh    = blockIdx.y;           // 0..15
  const int b     = bh >> 3;
  const int h     = bh & 7;
  const int qbase = (blockIdx.x * 4 + wave) * 16;
  const int half  = lane >> 4;
  const int col   = lane & 15;

  const _Float16* Qp = Qh + (size_t)bh * SQ * HD;
  const char*     Kb = (const char*)(Kh + (size_t)bh * SQ * HD);
  const char*     Vb = (const char*)(Vh + (size_t)bh * SQ * HD);
  const float*    mp = mask + (size_t)b * SQ;

  const unsigned ldsK_base = (unsigned)(size_t)(void*)ldsK;
  const unsigned ldsV_base = (unsigned)(size_t)(void*)ldsV;

  // Q A-fragments: 2 chunks over hd (contraction = head dim)
  v16h qa[2];
  {
    int qrow = qbase + col;
#pragma unroll
    for (int c = 0; c < 2; ++c)
#pragma unroll
      for (int i = 0; i < 8; ++i) {
        int k = 32 * c + 2 * i + 16 * half;
        qa[c][2 * i]     = Qp[(size_t)qrow * HD + k];
        qa[c][2 * i + 1] = Qp[(size_t)qrow * HD + k + 1];
      }
  }

  float mrun[8], lrun[8];
#pragma unroll
  for (int r = 0; r < 8; ++r) { mrun[r] = -1e30f; lrun[r] = 0.f; }
  v8f acc[4] = {};
  const float scale = 0.125f;             // 1/sqrt(64)

  for (int j = 0; j < SQ; j += 32) {
    // ---- stage K/V tile (32 keys x 64 dims, f16) into LDS, async ----
    // 32x64x2B = 4KB per tile -> 256 16B chunks; 128 threads x 2 chunks.
    {
      size_t tbyte = (size_t)j * HD * sizeof(_Float16);
#pragma unroll
      for (int u = 0; u < 2; ++u) {
        unsigned c = (unsigned)tid * 2u + u;        // chunk 0..255
        async_copy_b128(ldsK_base + c * 16u, Kb + tbyte + c * 16u);
        async_copy_b128(ldsV_base + c * 16u, Vb + tbyte + c * 16u);
      }
    }
    wait_async0();       // this wave's async copies landed in LDS
    __syncthreads();     // all waves' copies landed -> tile published

    // ---- scores: two 16x16 tiles (keys j..j+15, j+16..j+31) ----
    v8f sc[2];
#pragma unroll
    for (int t = 0; t < 2; ++t) {
      int kl = 16 * t + col;              // local key index 0..31
      v16h kb0, kb1;                      // B layout, contraction = head dim
#pragma unroll
      for (int i = 0; i < 8; ++i) {
        int d0 = 2 * i + 16 * half;
        kb0[2 * i]     = ldsK[kl * HD + d0];
        kb0[2 * i + 1] = ldsK[kl * HD + d0 + 1];
        kb1[2 * i]     = ldsK[kl * HD + 32 + d0];
        kb1[2 * i + 1] = ldsK[kl * HD + 32 + d0 + 1];
      }
      v8f c = {};
      c = wmma_f16(qa[0], kb0, c);
      c = wmma_f16(qa[1], kb1, c);
      float mv = mp[j + kl];
#pragma unroll
      for (int r = 0; r < 8; ++r) c[r] = c[r] * scale + mv;
      sc[t] = c;
    }

    // ---- online softmax: row max across the 16 lanes of each half ----
    float alpha[8];
#pragma unroll
    for (int r = 0; r < 8; ++r) {
      float v = fmaxf(sc[0][r], sc[1][r]);
#pragma unroll
      for (int m = 8; m >= 1; m >>= 1) v = fmaxf(v, __shfl_xor(v, m, 32));
      float mnew = fmaxf(mrun[r], v);
      alpha[r] = __expf(mrun[r] - mnew);
      mrun[r]  = mnew;
    }

    float rsum[8];
#pragma unroll
    for (int r = 0; r < 8; ++r) rsum[r] = 0.f;

#pragma unroll
    for (int t = 0; t < 2; ++t)
#pragma unroll
      for (int r = 0; r < 8; ++r) {
        float p = __expf(sc[t][r] - mrun[r]);
        rsum[r] += p;
        int row = r + 8 * half;
        ldsP[wave][row * 32 + 16 * t + col] = (_Float16)p;
      }

#pragma unroll
    for (int r = 0; r < 8; ++r) {
      float v = rsum[r];
#pragma unroll
      for (int m = 8; m >= 1; m >>= 1) v += __shfl_xor(v, m, 32);
      lrun[r] = lrun[r] * alpha[r] + v;
#pragma unroll
      for (int t = 0; t < 4; ++t) acc[t][r] *= alpha[r];
    }
    __syncthreads();     // P visible (RAW across lanes)

    // ---- reload P as A-fragment (contraction = key index, K=32) ----
    v16h pa;
    {
      int row = col;
#pragma unroll
      for (int i = 0; i < 8; ++i) {
        int k = 2 * i + 16 * half;
        pa[2 * i]     = ldsP[wave][row * 32 + k];
        pa[2 * i + 1] = ldsP[wave][row * 32 + k + 1];
      }
    }

    // ---- PV: 4 output-column tiles, contraction over 32 keys ----
#pragma unroll
    for (int t = 0; t < 4; ++t) {
      v16h vb;
#pragma unroll
      for (int i = 0; i < 8; ++i) {
        int kl = 2 * i + 16 * half;       // local key
        vb[2 * i]     = ldsV[kl * HD + 16 * t + col];
        vb[2 * i + 1] = ldsV[(kl + 1) * HD + 16 * t + col];
      }
      acc[t] = wmma_f16(pa, vb, acc[t]);
    }
    __syncthreads();     // all LDS reads done before next async staging
  }

  // ---- epilogue: normalize, store (B, S, nh*hd) f32 ----
#pragma unroll
  for (int r = 0; r < 8; ++r) {
    float rl = 1.0f / lrun[r];
    int   s  = qbase + r + 8 * half;
#pragma unroll
    for (int t = 0; t < 4; ++t) {
      int d = 16 * t + col;
      Out[((size_t)b * SQ + s) * (NH * HD) + h * HD + d] = acc[t][r] * rl;
    }
  }
}

// ---------------------------------------------------------------------------
extern "C" void kernel_launch(void* const* d_in, const int* in_sizes, int n_in,
                              void* d_out, int out_size, void* d_ws, size_t ws_size,
                              hipStream_t stream) {
  const float* X    = (const float*)d_in[0];
  const float* mask = (const float*)d_in[1];
  const float* Wq   = (const float*)d_in[2];
  const float* bq   = (const float*)d_in[3];
  const float* Wk   = (const float*)d_in[4];
  const float* bk   = (const float*)d_in[5];
  const float* Wv   = (const float*)d_in[6];
  const float* bv   = (const float*)d_in[7];
  float* out = (float*)d_out;

  const size_t per = (size_t)NB * NH * SQ * HD;   // 4M f16 elems each
  _Float16* Qh = (_Float16*)d_ws;
  _Float16* Kh = Qh + per;
  _Float16* Vh = Kh + per;

  // M = B*S = 8192 rows; 64 rows per block (4 waves x 16)
  qkv_rope_kernel<<<dim3(128, 8, 1), 128, 0, stream>>>(X, Wq, bq, Qh, 1);
  qkv_rope_kernel<<<dim3(128, 8, 1), 128, 0, stream>>>(X, Wk, bk, Kh, 1);
  qkv_rope_kernel<<<dim3(128, 8, 1), 128, 0, stream>>>(X, Wv, bv, Vh, 0);

  attn_kernel<<<dim3(SQ / 64, NB * NH, 1), 128, 0, stream>>>(Qh, Kh, Vh, mask, out);
}
